// DualGraphEncoder_29832842838182
// MI455X (gfx1250) — matrix operand order repaired and checked by reading, not compiled, not run on hardware
//
#include <hip/hip_runtime.h>

// ---------------------------------------------------------------------------
// Types for CDNA5 WMMA
// ---------------------------------------------------------------------------
typedef __attribute__((ext_vector_type(16))) _Float16 v16h;
typedef __attribute__((ext_vector_type(2)))  _Float16 h2;
typedef __attribute__((ext_vector_type(8)))  float    v8f;

#define NEG_SLOPE 0.2f

// ---------------------------------------------------------------------------
// Ordered-uint encoding of float for atomicMax-based segment max
// ---------------------------------------------------------------------------
__device__ __forceinline__ unsigned enc_f(float f) {
    unsigned u = __float_as_uint(f);
    return (u >> 31) ? ~u : (u | 0x80000000u);
}
__device__ __forceinline__ float dec_f(unsigned u) {
    return __uint_as_float((u >> 31) ? (u & 0x7FFFFFFFu) : ~u);
}
__device__ __forceinline__ float lrelu(float e) {
    return e >= 0.f ? e : NEG_SLOPE * e;
}

// ---------------------------------------------------------------------------
// K1: H = X(NxK) @ W(KxF), fp32 in/out, f16 WMMA compute, fp32 accumulate.
//
// Block = 128 threads (4 waves) computing a 16(M) x 64(N) strip:
//   - A tile (16x32) staged once per K-step into LDS, shared by all 4 waves
//     (float4-coalesced global loads, packed fp32->fp16 convert on store).
//   - B tile (32x64) staged transposed [n][k]: each thread handles k-pair x
//     n-quad units, so converts are v_cvt_pk_f16_f32 and LDS writes are
//     packed 2xf16 (ds_store_b32); fragment reads are contiguous K-pairs.
//   - K, F compile-time -> K-loop fully unrolled -> 12/4 static v_wmma.
// ---------------------------------------------------------------------------
template <int K, int F>
__global__ __launch_bounds__(128) void gemm_wmma_f16(
    const float* __restrict__ X, const float* __restrict__ W,
    float* __restrict__ H, int N)
{
    const int tm    = blockIdx.x;            // 16-row tile index
    const int nbase = blockIdx.y * 64;       // 64-col strip base
    const int t     = threadIdx.x;
    const int wave  = t >> 5;
    const int lane  = t & 31;
    const int half  = lane >> 4;             // 0: lanes 0-15, 1: lanes 16-31
    const int mrow  = lane & 15;

    __shared__ _Float16 aTile[16 * 32];      // [m][k], 1 KB
    __shared__ _Float16 bTile[64 * 32];      // [n][k] transposed, 4 KB
    h2* bt2 = (h2*)bTile;                    // packed (k even, k odd) pairs

    // A staging: 16 rows x 32 floats; thread -> one float4
    const int ar = t >> 3;                   // 0..15
    const int ac = (t & 7) * 4;              // 0,4,..,28
    const int arow_g = tm * 16 + ar;
    const int arow_c = arow_g < N ? arow_g : N - 1;   // clamp, zero via keep
    const float akeep = arow_g < N ? 1.f : 0.f;

    const int n_loc  = wave * 16 + (lane & 15);       // 0..63 within strip
    const int n_glob = nbase + n_loc;

    v8f acc = {};

#pragma unroll
    for (int k0 = 0; k0 < K; k0 += 32) {
        __syncthreads();                     // LDS reuse vs previous reads

        // ---- stage A (coalesced float4, packed convert to f16) ----
        const float4 av = *(const float4*)(X + (size_t)arow_c * K + k0 + ac);
        h2* as = (h2*)&aTile[ar * 32 + ac];
        as[0] = (h2){(_Float16)(akeep * av.x), (_Float16)(akeep * av.y)};
        as[1] = (h2){(_Float16)(akeep * av.z), (_Float16)(akeep * av.w)};

        // ---- stage B transposed: k-pair x n-quad units ----
        // 32k x 64n = 16 k-pairs x 16 n-quads = 256 units; 2 per thread.
        // Two coalesced b128 row loads, then 4 packed (k,k+1) b32 LDS writes.
#pragma unroll
        for (int j = 0; j < 2; ++j) {
            int unit = j * 128 + t;          // 0..255
            int k2   = unit >> 4;            // 0..15: k-pair index
            int nq   = (unit & 15) * 4;      // 0,4,..,60: n-quad base
            const float4 b0 =
                *(const float4*)(W + (size_t)(k0 + 2 * k2)     * F + nbase + nq);
            const float4 b1 =
                *(const float4*)(W + (size_t)(k0 + 2 * k2 + 1) * F + nbase + nq);
            bt2[((nq + 0) * 32 + 2 * k2) >> 1] = (h2){(_Float16)b0.x, (_Float16)b1.x};
            bt2[((nq + 1) * 32 + 2 * k2) >> 1] = (h2){(_Float16)b0.y, (_Float16)b1.y};
            bt2[((nq + 2) * 32 + 2 * k2) >> 1] = (h2){(_Float16)b0.z, (_Float16)b1.z};
            bt2[((nq + 3) * 32 + 2 * k2) >> 1] = (h2){(_Float16)b0.w, (_Float16)b1.w};
        }
        __syncthreads();

        // ---- build fragments from LDS (ISA 7.12.2 layouts) ----
        v16h a, b;
#pragma unroll
        for (int i = 0; i < 8; ++i) {
            int ka = ((i >> 2) << 4) + half * 8 + ((i & 3) << 1);  // A interleave
            a[2 * i]     = aTile[mrow * 32 + ka];
            a[2 * i + 1] = aTile[mrow * 32 + ka + 1];
            int kb = ((i >> 2) << 4) + half * 8 + ((i & 3) << 1);  // B interleave
            b[2 * i]     = bTile[n_loc * 32 + kb];
            b[2 * i + 1] = bTile[n_loc * 32 + kb + 1];
        }
        acc = __builtin_amdgcn_wmma_f32_16x16x32_f16(
            false, a, false, b, (short)0, acc, false, false);
    }

#pragma unroll
    for (int v = 0; v < 8; ++v) {
        int row = tm * 16 + v + half * 8;
        if (row < N)
            H[(size_t)row * F + n_glob] = acc[v];
    }
}

// ---------------------------------------------------------------------------
// K2: per-node attention scalars es = h.a_src, ed = h.a_dst.
// Also (re)initializes max / denom and zeroes the aggregation output.
// ---------------------------------------------------------------------------
__global__ void node_prep(
    const float* __restrict__ H, const float* __restrict__ avec,
    float* __restrict__ es, float* __restrict__ ed,
    unsigned* __restrict__ menc, float* __restrict__ den,
    float* __restrict__ Out, int N, int F)
{
    int i = blockIdx.x * blockDim.x + threadIdx.x;
    if (i >= N) return;
    const float* h = H + (size_t)i * F;
    float s0 = 0.f, s1 = 0.f;
    for (int f = 0; f < F; ++f) {
        float v = h[f];
        s0 += v * avec[f];
        s1 += v * avec[F + f];
    }
    es[i]   = s0;
    ed[i]   = s1;
    menc[i] = 0x007FFFFFu;     // enc_f(-inf)
    den[i]  = 0.f;
    float* o = Out + (size_t)i * F;
    for (int f = 0; f < F; ++f) o[f] = 0.f;
}

// ---------------------------------------------------------------------------
// K3: segment max over edges (+ self loops) via ordered-uint atomicMax.
// ---------------------------------------------------------------------------
__global__ void edge_max_k(
    const float* __restrict__ es, const float* __restrict__ ed,
    const long long* __restrict__ ei, long long E, int N,
    unsigned* __restrict__ menc)
{
    long long idx = (long long)blockIdx.x * blockDim.x + threadIdx.x;
    long long total = E + N;
    if (idx >= total) return;
    int s, d;
    if (idx < E) { s = (int)ei[idx]; d = (int)ei[E + idx]; }
    else         { s = d = (int)(idx - E); }
    float e = lrelu(es[s] + ed[d]);
    atomicMax(&menc[d], enc_f(e));
}

// ---------------------------------------------------------------------------
// K4: w = exp(e - max[dst]); denom[dst] += w; stash w per edge.
// ---------------------------------------------------------------------------
__global__ void edge_expsum_k(
    const float* __restrict__ es, const float* __restrict__ ed,
    const long long* __restrict__ ei, long long E, int N,
    const unsigned* __restrict__ menc, float* __restrict__ den,
    float* __restrict__ wbuf)
{
    long long idx = (long long)blockIdx.x * blockDim.x + threadIdx.x;
    long long total = E + N;
    if (idx >= total) return;
    int s, d;
    if (idx < E) { s = (int)ei[idx]; d = (int)ei[E + idx]; }
    else         { s = d = (int)(idx - E); }
    float e = lrelu(es[s] + ed[d]);
    float w = __expf(e - dec_f(menc[d]));
    wbuf[idx] = w;
    atomicAdd(&den[d], w);
}

// ---------------------------------------------------------------------------
// K5: Out[dst] += alpha * H[src]. One wave per edge; each lane owns a
// contiguous F/32 chunk -> one b128/b64 load + float atomics.
// ---------------------------------------------------------------------------
__global__ __launch_bounds__(256) void edge_aggregate_k(
    const float* __restrict__ H, const long long* __restrict__ ei,
    const float* __restrict__ wbuf, const float* __restrict__ den,
    long long E, int N, int F, float* __restrict__ Out)
{
    int lane = threadIdx.x & 31;
    long long widx = (long long)blockIdx.x * (blockDim.x >> 5) + (threadIdx.x >> 5);
    long long total = E + N;
    if (widx >= total) return;
    int s, d;
    if (widx < E) { s = (int)ei[widx]; d = (int)ei[E + widx]; }
    else          { s = d = (int)(widx - E); }
    float alpha = wbuf[widx] / (den[d] + 1e-16f);

    int per  = F >> 5;                         // 4 (F=128) or 2 (F=64)
    int base = lane * per;
    const float* hp = H   + (size_t)s * F + base;
    float*       op = Out + (size_t)d * F + base;
    __builtin_prefetch(hp, 0, 0);              // global_prefetch_b8

    if (per == 4) {
        const float4 v = *(const float4*)hp;   // global_load_b128
        atomicAdd(&op[0], alpha * v.x);
        atomicAdd(&op[1], alpha * v.y);
        atomicAdd(&op[2], alpha * v.z);
        atomicAdd(&op[3], alpha * v.w);
    } else {
        const float2 v = *(const float2*)hp;   // global_load_b64
        atomicAdd(&op[0], alpha * v.x);
        atomicAdd(&op[1], alpha * v.y);
    }
}

// ---------------------------------------------------------------------------
// K6: bias (+ optional ReLU)
// ---------------------------------------------------------------------------
__global__ void bias_act_k(float* __restrict__ Out, const float* __restrict__ b,
                           long long NF, int F, int relu)
{
    long long i = (long long)blockIdx.x * blockDim.x + threadIdx.x;
    if (i >= NF) return;
    int f = (int)(i % F);
    float v = Out[i] + b[f];
    if (relu) v = fmaxf(v, 0.f);
    Out[i] = v;
}

// ---------------------------------------------------------------------------
// K7: column mean over nodes -> 64-vector embedding. One block per feature.
// ---------------------------------------------------------------------------
__global__ void col_mean_k(const float* __restrict__ H, int N, int F,
                           float* __restrict__ emb)
{
    __shared__ float red[256];
    int f = blockIdx.x;
    float s = 0.f;
    for (int i = threadIdx.x; i < N; i += blockDim.x)
        s += H[(size_t)i * F + f];
    red[threadIdx.x] = s;
    __syncthreads();
    for (int off = 128; off > 0; off >>= 1) {
        if ((int)threadIdx.x < off) red[threadIdx.x] += red[threadIdx.x + off];
        __syncthreads();
    }
    if (threadIdx.x == 0) emb[f] = red[0] / (float)N;
}

// ---------------------------------------------------------------------------
// K8: normalize + cosine-style scores / temperature
// ---------------------------------------------------------------------------
__global__ void finalize_k(const float* __restrict__ emb,
                           const float* __restrict__ temp,
                           float* __restrict__ out, int S)
{
    int g = blockIdx.x * blockDim.x + threadIdx.x;
    if (g >= S) return;
    const float* p = emb;                       // persona embedding
    const float* s = emb + (size_t)(g + 1) * 64;
    float dot = 0.f, pn = 0.f, sn = 0.f;
    for (int f = 0; f < 64; ++f) {
        dot += p[f] * s[f];
        pn  += p[f] * p[f];
        sn  += s[f] * s[f];
    }
    out[g] = dot * rsqrtf(pn) * rsqrtf(sn) / temp[0];
}

// ---------------------------------------------------------------------------
// Host-side orchestration helpers
// ---------------------------------------------------------------------------
template <int K, int F>
static void run_gat_layer(const float* X, const long long* ei, long long E, int N,
                          const float* W, const float* avec,
                          const float* b, int relu,
                          float* Hbuf, float* Obuf,
                          float* es, float* ed, unsigned* menc, float* den,
                          float* wbuf, hipStream_t stream)
{
    int tilesM = (N + 15) / 16;
    dim3 grid((unsigned)tilesM, (unsigned)(F / 64));
    gemm_wmma_f16<K, F><<<grid, 128, 0, stream>>>(X, W, Hbuf, N);

    node_prep<<<(N + 255) / 256, 256, 0, stream>>>(Hbuf, avec, es, ed, menc,
                                                   den, Obuf, N, F);
    long long total = E + N;
    int eb = (int)((total + 255) / 256);
    edge_max_k<<<eb, 256, 0, stream>>>(es, ed, ei, E, N, menc);
    edge_expsum_k<<<eb, 256, 0, stream>>>(es, ed, ei, E, N, menc, den, wbuf);
    int ab = (int)((total + 7) / 8);
    edge_aggregate_k<<<ab, 256, 0, stream>>>(Hbuf, ei, wbuf, den, E, N, F, Obuf);
    long long nf = (long long)N * F;
    bias_act_k<<<(int)((nf + 255) / 256), 256, 0, stream>>>(Obuf, b, nf, F, relu);
}

static void run_graph(const float* X, const long long* ei, long long E, int N,
                      const float* W1, const float* a1, const float* b1,
                      const float* W2, const float* a2, const float* b2,
                      float* H1, float* HA, float* H2, float* HB,
                      float* es, float* ed, unsigned* menc, float* den,
                      float* wbuf, float* embOut, hipStream_t stream)
{
    run_gat_layer<384, 128>(X,  ei, E, N, W1, a1, b1, /*relu=*/1,
                            H1, HA, es, ed, menc, den, wbuf, stream);
    run_gat_layer<128, 64>(HA, ei, E, N, W2, a2, b2, /*relu=*/0,
                           H2, HB, es, ed, menc, den, wbuf, stream);
    col_mean_k<<<64, 256, 0, stream>>>(HB, N, 64, embOut);
}

extern "C" void kernel_launch(void* const* d_in, const int* in_sizes, int n_in,
                              void* d_out, int out_size, void* d_ws, size_t ws_size,
                              hipStream_t stream)
{
    (void)n_in; (void)out_size; (void)ws_size;

    const float*     persona_x  = (const float*)d_in[0];
    const long long* persona_ei = (const long long*)d_in[1];
    const float*     story_x    = (const float*)d_in[2];
    const long long* story_ei   = (const long long*)d_in[3];
    const float* p_W1 = (const float*)d_in[4];
    const float* p_a1 = (const float*)d_in[5];
    const float* p_b1 = (const float*)d_in[6];
    const float* p_W2 = (const float*)d_in[7];
    const float* p_a2 = (const float*)d_in[8];
    const float* p_b2 = (const float*)d_in[9];
    const float* s_W1 = (const float*)d_in[10];
    const float* s_a1 = (const float*)d_in[11];
    const float* s_b1 = (const float*)d_in[12];
    const float* s_W2 = (const float*)d_in[13];
    const float* s_a2 = (const float*)d_in[14];
    const float* s_b2 = (const float*)d_in[15];
    const float* temp = (const float*)d_in[16];

    const int IN = 384, HID = 128, OUTD = 64;
    int       N = in_sizes[0] / IN;
    long long E = (long long)in_sizes[1] / 2;
    int       S = in_sizes[2] / (N * IN);

    // Carve scratch from d_ws (reused sequentially across the 33 graphs)
    size_t off = 0;
    auto carve = [&](size_t bytes) -> void* {
        void* p = (void*)((char*)d_ws + off);
        off += (bytes + 255) & ~(size_t)255;
        return p;
    };
    float*    H1   = (float*)carve((size_t)N * HID  * sizeof(float));
    float*    HA   = (float*)carve((size_t)N * HID  * sizeof(float));
    float*    H2   = (float*)carve((size_t)N * OUTD * sizeof(float));
    float*    HB   = (float*)carve((size_t)N * OUTD * sizeof(float));
    float*    es   = (float*)carve((size_t)N * sizeof(float));
    float*    ed   = (float*)carve((size_t)N * sizeof(float));
    unsigned* menc = (unsigned*)carve((size_t)N * sizeof(unsigned));
    float*    den  = (float*)carve((size_t)N * sizeof(float));
    float*    wbuf = (float*)carve((size_t)(E + N) * sizeof(float));
    float*    emb  = (float*)carve((size_t)(S + 1) * OUTD * sizeof(float));

    // Persona graph -> emb[0]
    run_graph(persona_x, persona_ei, E, N,
              p_W1, p_a1, p_b1, p_W2, p_a2, p_b2,
              H1, HA, H2, HB, es, ed, menc, den, wbuf, emb, stream);

    // Story graphs (shared story weights) -> emb[1..S]
    for (int g = 0; g < S; ++g) {
        const float*     Xg  = story_x  + (size_t)g * (size_t)N * IN;
        const long long* eig = story_ei + (size_t)g * 2 * (size_t)E;
        run_graph(Xg, eig, E, N,
                  s_W1, s_a1, s_b1, s_W2, s_a2, s_b2,
                  H1, HA, H2, HB, es, ed, menc, den, wbuf,
                  emb + (size_t)(g + 1) * OUTD, stream);
    }

    finalize_k<<<1, 32, 0, stream>>>(emb, temp, (float*)d_out, S);
}
